// LocationSensitiveAttention_40759239639200
// MI455X (gfx1250) — matrix operand order repaired and checked by reading, not compiled, not run on hardware
//
#include <hip/hip_runtime.h>
#include <hip/hip_bf16.h>

// ---------------- problem constants ----------------
#define ENC_DIM   512
#define QUERY_DIM 1024
#define ATTN_DIM  128
#define CONV_CH   32
#define KERNEL_W  31
#define BATCH     64
#define SEQ       2048
#define TTILES    (SEQ / 16)     // 128
#define KTILES    (ENC_DIM / 32) // 16
#define NTILES    (ATTN_DIM / 16)// 8
#define NUM_TILES (BATCH * TTILES) // 8192

// workspace layout (dword offsets)
constexpr int OFF_Q    = 0;                   // 8192  floats : query [B,128]
constexpr int OFF_WK   = 8192;                // 32768 dwords : Wk  packed bf16 B-frags [kt][nt][lane][v]
constexpr int OFF_WLOC = 8192 + 32768;        // 2048  dwords : Wloc packed bf16 B-frags [nt][lane][v]
constexpr int OFF_LOC  = OFF_WLOC + 2048;     // 2097152 dwords : loc-feat packed bf16 A-frags [tile][lane][v]
constexpr int OFF_E    = OFF_LOC + 2097152;   // 131072 floats : energies [B,T]

// ---------------- types ----------------
typedef __bf16 v16bf __attribute__((ext_vector_type(16)));
typedef float  v8f   __attribute__((ext_vector_type(8)));

union Frag {
    v16bf v;
    __bf16 h[16];
    unsigned int u[8];
    uint4 q[2];
};

// native RNE convert via hardware cvt (no manual bit twiddling)
__device__ __forceinline__ unsigned int pack_bf16x2(float lo, float hi) {
    union { __bf16 h[2]; unsigned int u; } t;
    t.h[0] = (__bf16)lo;
    t.h[1] = (__bf16)hi;
    return t.u;
}

// ---------------- 1) query = decoder_hidden @ Wq ----------------
__global__ void lsa_query_kernel(const float* __restrict__ dh,
                                 const float* __restrict__ Wq,
                                 float* __restrict__ q) {
    int idx = blockIdx.x * blockDim.x + threadIdx.x;  // 8192
    int b = idx >> 7, a = idx & 127;
    const float* drow = dh + b * QUERY_DIM;
    float s = 0.f;
#pragma unroll 4
    for (int i = 0; i < QUERY_DIM; ++i) s += drow[i] * Wq[i * ATTN_DIM + a];
    q[idx] = s;
}

// ---------------- 2a) pack Wk into bf16 B-fragments ----------------
// B-matrix 32x16 bf16 layout: lane%16 = column N; half = lane/16;
// VGPR v holds K = half*8 + (v&3)*2 + (v>>2)*16 (lo), +1 (hi).
__global__ void lsa_pack_wk_kernel(const float* __restrict__ Wk,
                                   unsigned int* __restrict__ out) {
    int idx = blockIdx.x * blockDim.x + threadIdx.x;  // 32768
    int v = idx & 7, lane = (idx >> 3) & 31, nt = (idx >> 8) & 7, kt = idx >> 11;
    int n = nt * 16 + (lane & 15);
    int k = kt * 32 + (lane >> 4) * 8 + (v & 3) * 2 + (v >> 2) * 16;
    out[idx] = pack_bf16x2(Wk[k * ATTN_DIM + n], Wk[(k + 1) * ATTN_DIM + n]);
}

// ---------------- 2b) pack Wloc (32x128) into bf16 B-fragments ----------------
__global__ void lsa_pack_wloc_kernel(const float* __restrict__ Wloc,
                                     unsigned int* __restrict__ out) {
    int idx = blockIdx.x * blockDim.x + threadIdx.x;  // 2048
    int v = idx & 7, lane = (idx >> 3) & 31, nt = (idx >> 8) & 7;
    int n = nt * 16 + (lane & 15);
    int k = (lane >> 4) * 8 + (v & 3) * 2 + (v >> 2) * 16;
    out[idx] = pack_bf16x2(Wloc[k * ATTN_DIM + n], Wloc[(k + 1) * ATTN_DIM + n]);
}

// ---------------- 3) location conv -> bf16 A-fragments ----------------
// A-matrix 16x32 bf16 layout: lane%16 = row M (= t within tile); half = lane/16;
// VGPR v holds K = half*8 + (v&3)*2 + (v>>2)*16 (+1 in high half). K = conv channel.
__global__ void lsa_loc_kernel(const float* __restrict__ prev,
                               const float* __restrict__ cw,
                               const float* __restrict__ cb,
                               unsigned int* __restrict__ out) {
    int idx = blockIdx.x * blockDim.x + threadIdx.x;  // 2097152
    int v = idx & 7, lane = (idx >> 3) & 31, tt = (idx >> 8) & 127, b = idx >> 15;
    int t = tt * 16 + (lane & 15);
    int c0 = (lane >> 4) * 8 + (v & 3) * 2 + (v >> 2) * 16;
    float s0 = cb[c0], s1 = cb[c0 + 1];
    const float* pr = prev + b * SEQ;
#pragma unroll
    for (int j = 0; j < KERNEL_W; ++j) {
        int ts = t + j - (KERNEL_W - 1) / 2;
        float p = (ts >= 0 && ts < SEQ) ? pr[ts] : 0.f;
        s0 += p * cw[c0 * KERNEL_W + j];
        s1 += p * cw[(c0 + 1) * KERNEL_W + j];
    }
    out[idx] = pack_bf16x2(s0, s1);
}

// ---------------- 4) main fused kernel: WMMA GEMMs + tanh + v-dot ----------------
__global__ void __launch_bounds__(256) lsa_main_kernel(
    const float* __restrict__ enc, const float* __restrict__ query,
    const float* __restrict__ vvec,
    const unsigned int* __restrict__ wkp, const unsigned int* __restrict__ wlocp,
    const unsigned int* __restrict__ locp, float* __restrict__ energy) {

    const int lane = threadIdx.x & 31;
    const int wave = threadIdx.x >> 5;
    const int tile = blockIdx.x * 8 + wave;      // 0..8191
    const int b = tile >> 7, tt = tile & 127;
    const int row = lane & 15, half = lane >> 4;

    v8f acc[NTILES];
#pragma unroll
    for (int nt = 0; nt < NTILES; ++nt)
#pragma unroll
        for (int j = 0; j < 8; ++j) acc[nt][j] = 0.f;

    // --- location term: single K=32 WMMA per n-tile ---
    Frag afrag;
    {
        const uint4* lp = (const uint4*)(locp + (size_t)tile * 256 + lane * 8);
        afrag.q[0] = lp[0];
        afrag.q[1] = lp[1];
#pragma unroll
        for (int nt = 0; nt < NTILES; ++nt) {
            Frag bf;
            const uint4* wp = (const uint4*)(wlocp + (nt * 32 + lane) * 8);
            bf.q[0] = wp[0];
            bf.q[1] = wp[1];
            acc[nt] = __builtin_amdgcn_wmma_f32_16x16x32_bf16(
                false, afrag.v, false, bf.v, (short)0, acc[nt], false, false);
        }
    }

    // --- keys GEMM: encoder_outputs (fp32 stream -> bf16 via HW cvt) x Wk ---
    // Per k-step each lane covers K = k0..k0+7 and k0+16..k0+23 (k0 = kt*32 + half*8):
    // two aligned 32B chunks -> four float4 loads; 16 native bf16 casts fill the frag.
    const float4* Arow4 = (const float4*)(enc + ((size_t)b * SEQ + tt * 16) * ENC_DIM);
    const int abase = row * (ENC_DIM / 4) + half * 2;  // float4 units
    for (int kt = 0; kt < KTILES; ++kt) {
        const int i0 = abase + kt * 8;
        float4 c0 = Arow4[i0];
        float4 c1 = Arow4[i0 + 1];
        float4 c2 = Arow4[i0 + 4];
        float4 c3 = Arow4[i0 + 5];
        afrag.h[0]  = (__bf16)c0.x;  afrag.h[1]  = (__bf16)c0.y;
        afrag.h[2]  = (__bf16)c0.z;  afrag.h[3]  = (__bf16)c0.w;
        afrag.h[4]  = (__bf16)c1.x;  afrag.h[5]  = (__bf16)c1.y;
        afrag.h[6]  = (__bf16)c1.z;  afrag.h[7]  = (__bf16)c1.w;
        afrag.h[8]  = (__bf16)c2.x;  afrag.h[9]  = (__bf16)c2.y;
        afrag.h[10] = (__bf16)c2.z;  afrag.h[11] = (__bf16)c2.w;
        afrag.h[12] = (__bf16)c3.x;  afrag.h[13] = (__bf16)c3.y;
        afrag.h[14] = (__bf16)c3.z;  afrag.h[15] = (__bf16)c3.w;
#pragma unroll
        for (int nt = 0; nt < NTILES; ++nt) {
            Frag bf;
            const uint4* wp = (const uint4*)(wkp + ((kt * 8 + nt) * 32 + lane) * 8);
            bf.q[0] = wp[0];
            bf.q[1] = wp[1];
            acc[nt] = __builtin_amdgcn_wmma_f32_16x16x32_bf16(
                false, afrag.v, false, bf.v, (short)0, acc[nt], false, false);
        }
    }

    // --- epilogue: + query, tanh, dot with v, reduce over A ---
    float p[8];
#pragma unroll
    for (int r = 0; r < 8; ++r) p[r] = 0.f;
    const float* qrow = query + b * ATTN_DIM;
#pragma unroll
    for (int nt = 0; nt < NTILES; ++nt) {
        int a = nt * 16 + row;      // C/D layout: lane%16 = column N
        float qv = qrow[a];
        float vw = vvec[a];
#pragma unroll
        for (int r = 0; r < 8; ++r) {
            float x = acc[nt][r] + qv;
            float y;
            // gfx1250 TRANS op; v_nop satisfies the 1-op TRANS result hazard
            asm volatile("v_tanh_f32 %0, %1\n\tv_nop" : "=v"(y) : "v"(x));
            p[r] += y * vw;
        }
    }
    // reduce across the 16 lanes of each half (rows M = r + 8*half)
#pragma unroll
    for (int m = 1; m < 16; m <<= 1)
#pragma unroll
        for (int r = 0; r < 8; ++r) p[r] += __shfl_xor(p[r], m, 32);

    if (row == 0) {
        float* erow = energy + b * SEQ + tt * 16 + half * 8;
#pragma unroll
        for (int r = 0; r < 8; ++r) erow[r] = p[r];
    }
}

// ---------------- 5) softmax over T per batch row ----------------
__global__ void __launch_bounds__(256) lsa_softmax_kernel(const float* __restrict__ energy,
                                                          float* __restrict__ out) {
    __shared__ float red[256];
    const int b = blockIdx.x, t = threadIdx.x;
    const float* e = energy + b * SEQ;
    float local[8];
    float mx = -1e30f;
#pragma unroll
    for (int i = 0; i < 8; ++i) {
        local[i] = e[t + i * 256];
        mx = fmaxf(mx, local[i]);
    }
    red[t] = mx;
    __syncthreads();
    for (int s = 128; s > 0; s >>= 1) {
        if (t < s) red[t] = fmaxf(red[t], red[t + s]);
        __syncthreads();
    }
    mx = red[0];
    __syncthreads();
    float sum = 0.f;
#pragma unroll
    for (int i = 0; i < 8; ++i) {
        local[i] = __expf(local[i] - mx);
        sum += local[i];
    }
    red[t] = sum;
    __syncthreads();
    for (int s = 128; s > 0; s >>= 1) {
        if (t < s) red[t] += red[t + s];
        __syncthreads();
    }
    float inv = 1.f / red[0];
    float* o = out + b * SEQ;
#pragma unroll
    for (int i = 0; i < 8; ++i) o[t + i * 256] = local[i] * inv;
}

// ---------------- launcher ----------------
extern "C" void kernel_launch(void* const* d_in, const int* in_sizes, int n_in,
                              void* d_out, int out_size, void* d_ws, size_t ws_size,
                              hipStream_t stream) {
    const float* enc  = (const float*)d_in[0];   // [64,2048,512]
    const float* dh   = (const float*)d_in[1];   // [64,1024]
    const float* prev = (const float*)d_in[2];   // [64,2048]
    const float* Wq   = (const float*)d_in[3];   // [1024,128]
    const float* Wk   = (const float*)d_in[4];   // [512,128]
    const float* cw   = (const float*)d_in[5];   // [32,1,31]
    const float* cb   = (const float*)d_in[6];   // [32]
    const float* Wloc = (const float*)d_in[7];   // [32,128]
    const float* vv   = (const float*)d_in[8];   // [128]
    float* out = (float*)d_out;                  // [64,2048]

    float* ws = (float*)d_ws;
    float*        q     = ws + OFF_Q;
    unsigned int* wkp   = (unsigned int*)(ws + OFF_WK);
    unsigned int* wlocp = (unsigned int*)(ws + OFF_WLOC);
    unsigned int* locp  = (unsigned int*)(ws + OFF_LOC);
    float*        energ = ws + OFF_E;

    lsa_query_kernel    <<<32,   256, 0, stream>>>(dh, Wq, q);
    lsa_pack_wk_kernel  <<<128,  256, 0, stream>>>(Wk, wkp);
    lsa_pack_wloc_kernel<<<8,    256, 0, stream>>>(Wloc, wlocp);
    lsa_loc_kernel      <<<8192, 256, 0, stream>>>(prev, cw, cb, locp);
    lsa_main_kernel     <<<NUM_TILES / 8, 256, 0, stream>>>(enc, q, vv, wkp, wlocp, locp, energ);
    lsa_softmax_kernel  <<<BATCH, 256, 0, stream>>>(energ, out);
}
